// GNNnodeBased_17506286698688
// MI455X (gfx1250) — matrix operand order repaired and checked by reading, not compiled, not run on hardware
//
#include <hip/hip_runtime.h>
#include <hip/hip_bf16.h>
#include <math.h>

typedef __attribute__((ext_vector_type(16))) __bf16 v16bf;
typedef __attribute__((ext_vector_type(8)))  float  v8f;

#define BM 32
#define BN 64
#define BK 32
#define LDS_STRIDE 40   // BK + 8 bf16 pad: 20-dword row stride -> 16 rows hit 16 bank groups
#define THREADS 256
#define A_BUF (BM * LDS_STRIDE)   // elements per A buffer
#define B_BUF (BN * LDS_STRIDE)   // elements per B buffer

union FragBF { uint4 q[2]; v16bf v; };
union Pack4  { __bf16 h[4]; uint2 u2; };

// ---------------------------------------------------------------- utility

__global__ void k_zero_f(float* p, long n, const int* flag) {
    if (flag && flag[0] == 0) return;
    long i = (long)blockIdx.x * THREADS + threadIdx.x;
    if (i < n) p[i] = 0.f;
}

__global__ void k_zero_i(int* p, int n) {
    int i = blockIdx.x * THREADS + threadIdx.x;
    if (i < n) p[i] = 0;
}

// transpose-convert weights once: d[n*K + k] = s[k*Nc + n]; n in [0,NcPad), zero-pad n>=Nc
__global__ void k_cvt_bf16_t(const float* __restrict__ s, __bf16* __restrict__ d,
                             int K, int Nc, int NcPad) {
    long i = (long)blockIdx.x * THREADS + threadIdx.x;
    long total = (long)K * NcPad;
    if (i >= total) return;
    int n = (int)(i / K);
    int k = (int)(i - (long)n * K);
    d[i] = (n < Nc) ? (__bf16)s[(long)k * Nc + n] : (__bf16)0.f;
}

__global__ void k_init_state(const float* __restrict__ s0, float* __restrict__ state,
                             float* __restrict__ state_old, long n) {
    long i = (long)blockIdx.x * THREADS + threadIdx.x;
    if (i < n) { state[i] = s0[i]; state_old[i] = 1.f; }
}

// ---------------------------------------------------------------- SPMM scatter
// one wave (32 lanes) per edge; lane-strided features; f32 atomics into out.
__global__ void k_spmm(const float* __restrict__ vals, const int* __restrict__ src,
                       const int* __restrict__ dst, const float* __restrict__ dense,
                       long ld, long colOff, int D, float* __restrict__ out,
                       int E, const int* flag) {
    if (flag && flag[0] == 0) return;
    int e = blockIdx.x * (THREADS / 32) + (threadIdx.x >> 5);
    if (e >= E) return;
    int lane = threadIdx.x & 31;
    float v = vals[e];
    long srow = src ? (long)src[e] : (long)e;
    long drow = (long)dst[e];
    const float* in = dense + srow * ld + colOff;
    float* o = out + drow * (long)D;
    for (int j = lane; j < D; j += 32) atomicAdd(&o[j], v * in[j]);
}

// ---------------------------------------------------------------- convergence
__global__ void k_check(const float* __restrict__ state, const float* __restrict__ state_old,
                        int n, int* flag) {
    int i = blockIdx.x * THREADS + threadIdx.x;
    if (i >= n) return;
    const float4* s  = (const float4*)(state     + (long)i * 128);
    const float4* so = (const float4*)(state_old + (long)i * 128);
    float d2 = 0.f, n2 = 0.f;
    for (int j = 0; j < 32; ++j) {
        float4 a = s[j], b = so[j];
        float dx = a.x - b.x, dy = a.y - b.y, dz = a.z - b.z, dw = a.w - b.w;
        d2 += dx * dx + dy * dy + dz * dz + dw * dw;
        n2 += b.x * b.x + b.y * b.y + b.z * b.z + b.w * b.w;
    }
    if (d2 > (0.01f * 0.01f) * n2) atomicOr(flag, 1);
}

// ---------------------------------------------------------------- packing (vectorized x4)
// inp = concat[state(128), nodes(128), aggS(128), aggN(128), aggA(64)] as bf16
__global__ void k_pack_inp(const float* __restrict__ state, const float* __restrict__ nodes,
                           const float* __restrict__ aggS, const float* __restrict__ aggN,
                           const float* __restrict__ aggA, __bf16* __restrict__ inp,
                           int n, const int* flag) {
    if (flag[0] == 0) return;
    long idx = (long)blockIdx.x * THREADS + threadIdx.x;   // one 4-element chunk
    long total = (long)n * 144;                            // 576/4 chunks per row
    if (idx >= total) return;
    long row = idx / 144;
    int c = (int)(idx - row * 144) * 4;
    const float* src;
    if      (c < 128) src = state + row * 128 + c;
    else if (c < 256) src = nodes + row * 128 + (c - 128);
    else if (c < 384) src = aggS  + row * 128 + (c - 256);
    else if (c < 512) src = aggN  + row * 128 + (c - 384);
    else              src = aggA  + row * 64  + (c - 512);
    float4 v = *(const float4*)src;
    Pack4 o;
    o.h[0] = (__bf16)v.x; o.h[1] = (__bf16)v.y; o.h[2] = (__bf16)v.z; o.h[3] = (__bf16)v.w;
    *(uint2*)(inp + row * 576 + c) = o.u2;
}

// ho = concat[state(128), nodes(128)] as bf16
__global__ void k_pack_out(const float* __restrict__ state, const float* __restrict__ nodes,
                           __bf16* __restrict__ ho, int n) {
    long idx = (long)blockIdx.x * THREADS + threadIdx.x;
    long total = (long)n * 64;                             // 256/4 chunks per row
    if (idx >= total) return;
    long row = idx / 64;
    int c = (int)(idx - row * 64) * 4;
    const float* src = (c < 128) ? (state + row * 128 + c) : (nodes + row * 128 + (c - 128));
    float4 v = *(const float4*)src;
    Pack4 o;
    o.h[0] = (__bf16)v.x; o.h[1] = (__bf16)v.y; o.h[2] = (__bf16)v.z; o.h[3] = (__bf16)v.w;
    *(uint2*)(ho + row * 256 + c) = o.u2;
}

// ---------------------------------------------------------------- WMMA GEMM core
// A: MxK bf16 row-major. Bt: NcPad x K bf16 (pre-transposed, n-major).
// Block tile 32x64, 8 waves, one v_wmma_f32_16x16x32_bf16 per wave per K-step.
// Software-pipelined: GLOBAL_LOAD_ASYNC_TO_LDS_B128 streams tile k+1 into the
// other LDS buffer while tile k is computed; in-order ASYNCcnt lets us wait
// only for the older copies (waves 0-3 issue A+B -> wait <=2; waves 4-7 issue
// B -> wait <=1). Out-of-range rows/cols are CLAMPED (padding rows of A only
// feed padding rows of C; padding cols of Bt only feed never-stored C cols).
__device__ __forceinline__ v8f wmma_tile_bf16(
    const __bf16* __restrict__ A, const __bf16* __restrict__ Bt,
    int M, int K, int NcPad, int blockM, int blockN,
    __bf16* sA, __bf16* sB)
{
    const int tid  = threadIdx.x;
    const int lane = tid & 31;
    const int wave = tid >> 5;
    const int wm = wave >> 2;   // 0..1
    const int wn = wave & 3;    // 0..3

    // staging: one 16-byte async copy per thread per tile, addresses fixed across K
    const int ar = tid >> 2;            // A row 0..31 (tid<128 only)
    const int ac = (tid & 3) * 8;       // 8-bf16 chunk within 32-wide tile row
    int gm = blockM + ar; if (gm > M - 1) gm = M - 1;
    const __bf16* aSrc = A + (long)gm * K + ac;
    const unsigned aDst0 = (unsigned)(size_t)(sA + ar * LDS_STRIDE + ac);
    const unsigned aDst1 = aDst0 + (unsigned)(A_BUF * 2);   // byte offset of buffer 1

    const int bn = tid >> 2;            // Bt row (column of B) 0..63
    int gn = blockN + bn; if (gn > NcPad - 1) gn = NcPad - 1;
    const __bf16* bSrc = Bt + (long)gn * K + ac;
    const unsigned bDst0 = (unsigned)(size_t)(sB + bn * LDS_STRIDE + ac);
    const unsigned bDst1 = bDst0 + (unsigned)(B_BUF * 2);

    const int kh = lane >> 4;
    const __bf16* arow0 = sA + (wm * 16 + (lane & 15)) * LDS_STRIDE;
    const __bf16* brow0 = sB + (wn * 16 + (lane & 15)) * LDS_STRIDE;

    // prologue: stage tile 0 into buffer 0
    if (tid < 128)
        asm volatile("global_load_async_to_lds_b128 %0, %1, off"
                     :: "v"(aDst0), "v"(aSrc) : "memory");
    asm volatile("global_load_async_to_lds_b128 %0, %1, off"
                 :: "v"(bDst0), "v"(bSrc) : "memory");

    v8f acc = {};
    int parity = 0;
    for (int k0 = 0; k0 < K; k0 += BK) {
        // issue async copy of the NEXT tile into the other buffer
        // (last step re-copies its own tile into the dead buffer: no OOB, no branch)
        long knext = (k0 + BK < K) ? (long)(k0 + BK) : (long)k0;
        if (tid < 128)
            asm volatile("global_load_async_to_lds_b128 %0, %1, off"
                         :: "v"(parity ? aDst0 : aDst1), "v"(aSrc + knext) : "memory");
        asm volatile("global_load_async_to_lds_b128 %0, %1, off"
                     :: "v"(parity ? bDst0 : bDst1), "v"(bSrc + knext) : "memory");
        __builtin_prefetch(bSrc + knext + BK, 0, 3);
        // wait only for the OLDER copies (in-order completion): newest 2 (A+B) or 1 (B)
        if (tid < 128) asm volatile("s_wait_asynccnt 0x2" ::: "memory");
        else           asm volatile("s_wait_asynccnt 0x1" ::: "memory");
        __syncthreads();
        const __bf16* arow = parity ? (arow0 + A_BUF) : arow0;
        const __bf16* brow = parity ? (brow0 + B_BUF) : brow0;
        FragBF fa, fb;
        // A 16x32 bf16 layout: lanes 0-15 -> K {0..7,16..23}; lanes 16-31 -> K {8..15,24..31}
        fa.q[0] = *(const uint4*)(arow + kh * 8);
        fa.q[1] = *(const uint4*)(arow + 16 + kh * 8);
        // B 32x16 bf16 layout: same K order as A: lanes 0-15 -> K {0..7,16..23}; lanes 16-31 -> K {8..15,24..31}
        fb.q[0] = *(const uint4*)(brow + kh * 8);
        fb.q[1] = *(const uint4*)(brow + 16 + kh * 8);
        acc = __builtin_amdgcn_wmma_f32_16x16x32_bf16(false, fa.v, false, fb.v,
                                                      (short)0, acc, false, false);
        __syncthreads();   // all waves done reading this buffer before it is refilled
        parity ^= 1;
    }
    return acc;
}

// tanh(A@B + bias) -> bf16 output (Nc multiple of 64 here)
__global__ __launch_bounds__(THREADS)
void k_gemm_tanh_bf16(const __bf16* __restrict__ A, const __bf16* __restrict__ Bt,
                      const float* __restrict__ bias, __bf16* __restrict__ out,
                      int M, int K, int Nc, const int* flag) {
    if (flag && flag[0] == 0) return;
    __shared__ __align__(16) __bf16 sA[2 * A_BUF];
    __shared__ __align__(16) __bf16 sB[2 * B_BUF];
    int blockM = blockIdx.y * BM, blockN = blockIdx.x * BN;
    v8f acc = wmma_tile_bf16(A, Bt, M, K, Nc, blockM, blockN, sA, sB);
    int lane = threadIdx.x & 31, wave = threadIdx.x >> 5;
    int wm = wave >> 2, wn = wave & 3, half = lane >> 4;
    int gn = blockN + wn * 16 + (lane & 15);
    for (int r = 0; r < 8; ++r) {
        int gm = blockM + wm * 16 + half * 8 + r;
        if (gm < M)
            out[(long)gm * Nc + gn] = (__bf16)tanhf(acc[r] + bias[gn]);
    }
}

// state update GEMM: state_old <- state; state <- tanh(A@B + bias). Nc = 128.
__global__ __launch_bounds__(THREADS)
void k_gemm_state(const __bf16* __restrict__ A, const __bf16* __restrict__ Bt,
                  const float* __restrict__ bias, float* __restrict__ state,
                  float* __restrict__ state_old, int M, int K, const int* flag) {
    if (flag[0] == 0) return;
    __shared__ __align__(16) __bf16 sA[2 * A_BUF];
    __shared__ __align__(16) __bf16 sB[2 * B_BUF];
    int blockM = blockIdx.y * BM, blockN = blockIdx.x * BN;
    v8f acc = wmma_tile_bf16(A, Bt, M, K, 128, blockM, blockN, sA, sB);
    int lane = threadIdx.x & 31, wave = threadIdx.x >> 5;
    int wm = wave >> 2, wn = wave & 3, half = lane >> 4;
    int gn = blockN + wn * 16 + (lane & 15);
    for (int r = 0; r < 8; ++r) {
        int gm = blockM + wm * 16 + half * 8 + r;
        if (gm < M) {
            long idx = (long)gm * 128 + gn;
            float old = state[idx];
            state_old[idx] = old;
            state[idx] = tanhf(acc[r] + bias[gn]);
        }
    }
}

// output head GEMM: out[gm,gn<7] = (A@B + bias) * mask; Bt padded to 16 cols
__global__ __launch_bounds__(THREADS)
void k_gemm_out(const __bf16* __restrict__ A, const __bf16* __restrict__ Bt,
                const float* __restrict__ bias, const unsigned char* __restrict__ m1,
                const unsigned char* __restrict__ m2, float* __restrict__ out,
                int M, int K) {
    __shared__ __align__(16) __bf16 sA[2 * A_BUF];
    __shared__ __align__(16) __bf16 sB[2 * B_BUF];
    int blockM = blockIdx.y * BM, blockN = blockIdx.x * BN;
    v8f acc = wmma_tile_bf16(A, Bt, M, K, 16, blockM, blockN, sA, sB);
    int lane = threadIdx.x & 31, wave = threadIdx.x >> 5;
    int wm = wave >> 2, wn = wave & 3, half = lane >> 4;
    int gn = blockN + wn * 16 + (lane & 15);
    for (int r = 0; r < 8; ++r) {
        int gm = blockM + wm * 16 + half * 8 + r;
        if (gm < M && gn < 7) {
            float mk = (m1[gm] && m2[gm]) ? 1.f : 0.f;
            out[(long)gm * 7 + gn] = (acc[r] + bias[gn]) * mk;
        }
    }
}

// ---------------------------------------------------------------- launch

extern "C" void kernel_launch(void* const* d_in, const int* in_sizes, int n_in,
                              void* d_out, int out_size, void* d_ws, size_t ws_size,
                              hipStream_t stream) {
    const float* nodes      = (const float*)d_in[0];
    const float* arcs       = (const float*)d_in[1];
    const unsigned char* set_mask = (const unsigned char*)d_in[2];
    const unsigned char* out_mask = (const unsigned char*)d_in[3];
    const int*   adj_src    = (const int*)d_in[4];
    const int*   adj_dst    = (const int*)d_in[5];
    const float* adj_vals   = (const float*)d_in[6];
    const int*   an_dst     = (const int*)d_in[7];
    const float* an_vals    = (const float*)d_in[8];
    const float* state_init = (const float*)d_in[9];
    const float* Ws1 = (const float*)d_in[10]; const float* bs1 = (const float*)d_in[11];
    const float* Ws2 = (const float*)d_in[12]; const float* bs2 = (const float*)d_in[13];
    const float* Wo1 = (const float*)d_in[14]; const float* bo1 = (const float*)d_in[15];
    const float* Wo2 = (const float*)d_in[16]; const float* bo2 = (const float*)d_in[17];
    float* out = (float*)d_out;

    const int N = in_sizes[0] / 128;
    const int E = in_sizes[4];
    const int MAX_IT = 10;

    // workspace carve-up (256B aligned)
    char* w = (char*)d_ws;
    size_t off = 0;
    auto take = [&](size_t bytes) -> void* {
        void* p = w + off; off = (off + bytes + 255) & ~(size_t)255; return p;
    };
    __bf16* Ws1t   = (__bf16*)take((size_t)512 * 576 * 2);   // [512][576] n-major
    __bf16* Ws2t   = (__bf16*)take((size_t)128 * 512 * 2);   // [128][512]
    __bf16* Wo1t   = (__bf16*)take((size_t)512 * 256 * 2);   // [512][256]
    __bf16* Wo2t   = (__bf16*)take((size_t)16  * 512 * 2);   // [16][512], cols 7..15 zero
    float*  aggA   = (float*)take((size_t)N * 64  * 4);
    float*  aggN   = (float*)take((size_t)N * 128 * 4);
    float*  aggS   = (float*)take((size_t)N * 128 * 4);
    float*  state  = (float*)take((size_t)N * 128 * 4);
    float*  stateO = (float*)take((size_t)N * 128 * 4);
    __bf16* inp    = (__bf16*)take((size_t)N * 576 * 2);   // also reused as ho
    __bf16* hidden = (__bf16*)take((size_t)N * 512 * 2);   // also reused as tbuf
    int*    flags  = (int*)take(64);

    auto blocks1d = [](long n) { return (unsigned)((n + THREADS - 1) / THREADS); };

    // setup: flags, transposed bf16 weights, static aggregations, state init
    k_zero_i<<<1, THREADS, 0, stream>>>(flags, 16);
    k_cvt_bf16_t<<<blocks1d((long)576 * 512), THREADS, 0, stream>>>(Ws1, Ws1t, 576, 512, 512);
    k_cvt_bf16_t<<<blocks1d((long)512 * 128), THREADS, 0, stream>>>(Ws2, Ws2t, 512, 128, 128);
    k_cvt_bf16_t<<<blocks1d((long)256 * 512), THREADS, 0, stream>>>(Wo1, Wo1t, 256, 512, 512);
    k_cvt_bf16_t<<<blocks1d((long)512 * 16),  THREADS, 0, stream>>>(Wo2, Wo2t, 512, 7, 16);
    k_zero_f<<<blocks1d((long)N * 64),  THREADS, 0, stream>>>(aggA, (long)N * 64,  nullptr);
    k_zero_f<<<blocks1d((long)N * 128), THREADS, 0, stream>>>(aggN, (long)N * 128, nullptr);
    {
        unsigned gb = (unsigned)((E + 7) / 8);
        k_spmm<<<gb, THREADS, 0, stream>>>(an_vals, nullptr, an_dst, arcs, 66, 2, 64, aggA, E, nullptr);
        k_spmm<<<gb, THREADS, 0, stream>>>(adj_vals, adj_src, adj_dst, nodes, 128, 0, 128, aggN, E, nullptr);
    }
    k_init_state<<<blocks1d((long)N * 128), THREADS, 0, stream>>>(state_init, state, stateO, (long)N * 128);

    // fixed-point iterations, device-side convergence gate
    dim3 g1((512 + BN - 1) / BN, (N + BM - 1) / BM);   // 576 -> 512
    dim3 g2((128 + BN - 1) / BN, (N + BM - 1) / BM);   // 512 -> 128
    for (int it = 0; it < MAX_IT; ++it) {
        int* flag = flags + it;
        k_check<<<blocks1d(N), THREADS, 0, stream>>>(state, stateO, N, flag);
        k_zero_f<<<blocks1d((long)N * 128), THREADS, 0, stream>>>(aggS, (long)N * 128, flag);
        k_spmm<<<(unsigned)((E + 7) / 8), THREADS, 0, stream>>>(adj_vals, adj_src, adj_dst,
                                                                state, 128, 0, 128, aggS, E, flag);
        k_pack_inp<<<blocks1d((long)N * 144), THREADS, 0, stream>>>(state, nodes, aggS, aggN, aggA,
                                                                    inp, N, flag);
        k_gemm_tanh_bf16<<<g1, THREADS, 0, stream>>>(inp, Ws1t, bs1, hidden, N, 576, 512, flag);
        k_gemm_state<<<g2, THREADS, 0, stream>>>(hidden, Ws2t, bs2, state, stateO, N, 512, flag);
    }

    // output head
    __bf16* ho   = inp;      // N x 256 bf16 (reuse)
    __bf16* tbuf = hidden;   // N x 512 bf16 (reuse)
    k_pack_out<<<blocks1d((long)N * 64), THREADS, 0, stream>>>(state, nodes, ho, N);
    dim3 g3((512 + BN - 1) / BN, (N + BM - 1) / BM);
    k_gemm_tanh_bf16<<<g3, THREADS, 0, stream>>>(ho, Wo1t, bo1, tbuf, N, 256, 512, nullptr);
    dim3 g4(1, (N + BM - 1) / BM);
    k_gemm_out<<<g4, THREADS, 0, stream>>>(tbuf, Wo2t, bo2, set_mask, out_mask, out, N, 512);
}
